// AttentionBlock_71287867179126
// MI455X (gfx1250) — matrix operand order repaired and checked by reading, not compiled, not run on hardware
//
#include <hip/hip_runtime.h>
#include <hip/hip_bf16.h>

typedef _Float16 half_t;
typedef __attribute__((ext_vector_type(16))) _Float16 v16h;
typedef __attribute__((ext_vector_type(8)))  _Float16 v8h;
typedef __attribute__((ext_vector_type(8)))  float    v8f;

#define B_   8
#define C_   512
#define N_   1024
#define HEADS_ 8
#define HD_  64

__device__ __forceinline__ v16h combine16(v8h lo, v8h hi) {
    v16h r;
#pragma unroll
    for (int i = 0; i < 8; ++i) { r[i] = lo[i]; r[i + 8] = hi[i]; }
    return r;
}

// A fragment: 16x32 (MxK) f16, src points at tile element (m=0,k=0), row stride ld.
// Layout per ISA: lanes 0-15 hold M=lane, K chunks [0..7] and [16..23];
// lanes 16-31 hold M=lane-16, K chunks [8..15] and [24..31].
__device__ __forceinline__ v16h load_a_frag(const half_t* src, int ld) {
    int lane = threadIdx.x & 31;
    int r = lane & 15, h = lane >> 4;
    const half_t* p = src + (size_t)r * ld + h * 8;
    v8h lo = *(const v8h*)(p);
    v8h hi = *(const v8h*)(p + 16);
    return combine16(lo, hi);
}

// B fragment: 32x16 (KxN) f16 from an N-major source (column n has contiguous K),
// &B[k][n] = src + n*ld + k. Lanes 0-15: N=lane, K [0..7] and [16..23]; lanes 16-31: N=lane-16, K [8..15] and [24..31].
__device__ __forceinline__ v16h load_b_frag(const half_t* src, int ld) {
    int lane = threadIdx.x & 31;
    int n = lane & 15, h = lane >> 4;
    const half_t* p = src + (size_t)n * ld + h * 8;
    v8h lo = *(const v8h*)(p);
    v8h hi = *(const v8h*)(p + 16);
    return combine16(lo, hi);
}

__device__ __forceinline__ v8f wmma_f16(v16h a, v16h b, v8f c) {
    return __builtin_amdgcn_wmma_f32_16x16x32_f16(false, a, false, b, (short)0, c, false, false);
}

// ---------------- GroupNorm ----------------

__global__ void gn_stats_kernel(const float* __restrict__ x, float* __restrict__ stats) {
    int gid = blockIdx.x;                        // b*32 + g
    const float* p = x + (size_t)gid * 16 * N_;  // 16 channels * 1024 spatial, contiguous
    float s = 0.f, ss = 0.f;
    for (int i = threadIdx.x; i < 16 * N_; i += 256) {
        float v = p[i];
        s += v; ss += v * v;
    }
    __shared__ float sh0[256], sh1[256];
    sh0[threadIdx.x] = s; sh1[threadIdx.x] = ss;
    __syncthreads();
    for (int off = 128; off > 0; off >>= 1) {
        if ((int)threadIdx.x < off) {
            sh0[threadIdx.x] += sh0[threadIdx.x + off];
            sh1[threadIdx.x] += sh1[threadIdx.x + off];
        }
        __syncthreads();
    }
    if (threadIdx.x == 0) {
        float inv = 1.0f / (16.0f * N_);
        float mean = sh0[0] * inv;
        float var  = sh1[0] * inv - mean * mean;
        stats[2 * gid]     = mean;
        stats[2 * gid + 1] = rsqrtf(var + 1e-5f);
    }
}

// Normalize + affine + cast to f16, transposing (b,c,s) -> (b,s,c)
__global__ void gn_apply_kernel(const float* __restrict__ x,
                                const float* __restrict__ scale,
                                const float* __restrict__ bias,
                                const float* __restrict__ stats,
                                half_t* __restrict__ xn) {
    int t = blockIdx.x * 256 + threadIdx.x;      // (b, s) token index, 0..8191
    int b = t >> 10, s = t & 1023;
    const float* xb = x + (size_t)b * C_ * N_;
    half_t* dst = xn + (size_t)t * C_;
    for (int c0 = 0; c0 < C_; c0 += 8) {
        v8h out;
#pragma unroll
        for (int i = 0; i < 8; ++i) {
            int c = c0 + i;
            int g = c >> 4;
            float mean = stats[2 * (b * 32 + g)];
            float rstd = stats[2 * (b * 32 + g) + 1];
            float v = (xb[(size_t)c * N_ + s] - mean) * rstd * scale[c] + bias[c];
            out[i] = (half_t)v;
        }
        *(v8h*)(dst + c0) = out;
    }
}

// ---------------- weight cast fp32 -> f16 ----------------

__global__ void cast_w_kernel(const float* __restrict__ w, half_t* __restrict__ out, int n) {
    int i = blockIdx.x * 256 + threadIdx.x;
    if (i < n) out[i] = (half_t)w[i];
}

// ---------------- QKV projection GEMM ----------------
// out[m, o] = sum_c xn[m, c] * W[o, c] + bias[o]; M = B*N = 8192, K = N(out) = 512.
// Block: 128x64 output tile, 8 waves as 4x2 of 32x32 wave tiles.
__global__ __launch_bounds__(256)
void qkv_kernel(const half_t* __restrict__ xn, const half_t* __restrict__ w16,
                const float* __restrict__ bq, const float* __restrict__ bk,
                const float* __restrict__ bv,
                half_t* __restrict__ q16, half_t* __restrict__ k16,
                half_t* __restrict__ vt16) {
    int which = blockIdx.z;
    const half_t* W = w16 + (size_t)which * C_ * C_;
    const float* bias = (which == 0) ? bq : ((which == 1) ? bk : bv);

    int Mb = blockIdx.y * 128, Nb = blockIdx.x * 64;
    int w  = threadIdx.x >> 5;
    int wm = w >> 1, wn = w & 1;
    int mBase = Mb + wm * 32, nBase = Nb + wn * 32;

    v8f acc[2][2] = {};
    for (int kb = 0; kb < C_; kb += 32) {
        if (kb + 32 < C_)
            __builtin_prefetch(xn + (size_t)mBase * C_ + kb + 32, 0, 1);
        v16h a0 = load_a_frag(xn + (size_t)mBase * C_ + kb, C_);
        v16h a1 = load_a_frag(xn + (size_t)(mBase + 16) * C_ + kb, C_);
        v16h b0 = load_b_frag(W + (size_t)nBase * C_ + kb, C_);
        v16h b1 = load_b_frag(W + (size_t)(nBase + 16) * C_ + kb, C_);
        acc[0][0] = wmma_f16(a0, b0, acc[0][0]);
        acc[0][1] = wmma_f16(a0, b1, acc[0][1]);
        acc[1][0] = wmma_f16(a1, b0, acc[1][0]);
        acc[1][1] = wmma_f16(a1, b1, acc[1][1]);
    }

    int lane = threadIdx.x & 31, lr = lane & 15, lh = lane >> 4;
#pragma unroll
    for (int i = 0; i < 2; ++i)
#pragma unroll
        for (int j = 0; j < 2; ++j) {
            int col = nBase + j * 16 + lr;
            float bc = bias[col];
#pragma unroll
            for (int r = 0; r < 8; ++r) {
                int row = mBase + i * 16 + r + lh * 8;   // token index b*1024+s
                float v = acc[i][j][r] + bc;
                if (which < 2) {
                    half_t* dst = (which == 0) ? q16 : k16;
                    dst[(size_t)row * C_ + col] = (half_t)v;
                } else {
                    int b = row >> 10, s = row & 1023;
                    // V stored transposed: (b, h, d, m)
                    vt16[((size_t)(b * HEADS_ + (col >> 6)) * HD_ + (col & 63)) * N_ + s] = (half_t)v;
                }
            }
        }
}

// ---------------- Flash attention ----------------
// grid: (N/128, B*HEADS). Each wave owns 16 query rows; streams K/V in 32-row blocks.
__global__ __launch_bounds__(256)
void attn_kernel(const half_t* __restrict__ q16, const half_t* __restrict__ k16,
                 const half_t* __restrict__ vt16, half_t* __restrict__ o16) {
    int bh = blockIdx.y;
    int b = bh >> 3, h = bh & 7;
    int w = threadIdx.x >> 5, lane = threadIdx.x & 31, lr = lane & 15, lh = lane >> 4;
    int qrow = blockIdx.x * 128 + w * 16;

    __shared__ __align__(16) half_t kT[32 * 64];       // K block: 32 rows (m) x 64 (d)
    __shared__ __align__(16) half_t vT[64 * 32];       // V^T block: 64 rows (d) x 32 (m)
    __shared__ __align__(16) half_t pB[8 * 16 * 32];   // per-wave P staging

    // Q fragments, pre-scaled by 1/sqrt(64) = 0.125 (exact in f16)
    const half_t* qp = q16 + ((size_t)(b * N_ + qrow)) * C_ + h * HD_;
    v16h qf0 = load_a_frag(qp, C_);
    v16h qf1 = load_a_frag(qp + 32, C_);
#pragma unroll
    for (int i = 0; i < 16; ++i) {
        qf0[i] = qf0[i] * (half_t)0.125f;
        qf1[i] = qf1[i] * (half_t)0.125f;
    }

    v8f oacc[4] = {};
    float mrun[8], lrun[8];
#pragma unroll
    for (int r = 0; r < 8; ++r) { mrun[r] = -1e30f; lrun[r] = 0.f; }
    half_t* myP = pB + w * 16 * 32;

    for (int mb = 0; mb < N_; mb += 32) {
        {   // cooperative stage of K and V^T tiles
            int t = threadIdx.x;
            int kr = t >> 3, ko = (t & 7) * 8;
            *(v8h*)(kT + kr * 64 + ko) =
                *(const v8h*)(k16 + ((size_t)(b * N_ + mb + kr)) * C_ + h * HD_ + ko);
            int vd = t >> 2, vo = (t & 3) * 8;
            *(v8h*)(vT + vd * 32 + vo) =
                *(const v8h*)(vt16 + ((size_t)(bh * HD_ + vd)) * N_ + mb + vo);
        }
        __syncthreads();

        // S tile 16x32 = Q (16x64) * K^T (64x32), two 16-col WMMA tiles, k-split 32+32
        v8f s0 = {}, s1 = {};
        s0 = wmma_f16(qf0, load_b_frag(kT, 64), s0);
        s0 = wmma_f16(qf1, load_b_frag(kT + 32, 64), s0);
        s1 = wmma_f16(qf0, load_b_frag(kT + 16 * 64, 64), s1);
        s1 = wmma_f16(qf1, load_b_frag(kT + 16 * 64 + 32, 64), s1);

        // streaming softmax update; each C-matrix row lives in 16 lanes of a half-wave
#pragma unroll
        for (int r = 0; r < 8; ++r) {
            float a = s0[r], c = s1[r];
            float t = fmaxf(a, c);
            t = fmaxf(t, __shfl_xor(t, 1));
            t = fmaxf(t, __shfl_xor(t, 2));
            t = fmaxf(t, __shfl_xor(t, 4));
            t = fmaxf(t, __shfl_xor(t, 8));
            float mnew = fmaxf(mrun[r], t);
            float sc = __expf(mrun[r] - mnew);
            float pa = __expf(a - mnew), pc = __expf(c - mnew);
            float ps = pa + pc;
            ps += __shfl_xor(ps, 1);
            ps += __shfl_xor(ps, 2);
            ps += __shfl_xor(ps, 4);
            ps += __shfl_xor(ps, 8);
            lrun[r] = lrun[r] * sc + ps;
            mrun[r] = mnew;
#pragma unroll
            for (int tt = 0; tt < 4; ++tt) oacc[tt][r] *= sc;
            // C-layout -> LDS (row-major 16x32) for A-fragment reload
            int prow = r + lh * 8;
            myP[prow * 32 + lr]      = (half_t)pa;
            myP[prow * 32 + 16 + lr] = (half_t)pc;
        }
        v16h pf = load_a_frag(myP, 32);

        // O += P (16x32) * V (32x64): four 16-col WMMA tiles from V^T in LDS
#pragma unroll
        for (int tt = 0; tt < 4; ++tt)
            oacc[tt] = wmma_f16(pf, load_b_frag(vT + tt * 16 * 32, 32), oacc[tt]);

        __syncthreads();
    }

    half_t* op = o16 + ((size_t)(b * N_ + qrow)) * C_ + h * HD_;
#pragma unroll
    for (int tt = 0; tt < 4; ++tt) {
#pragma unroll
        for (int r = 0; r < 8; ++r) {
            float v = oacc[tt][r] / lrun[r];
            op[(size_t)(r + lh * 8) * C_ + tt * 16 + lr] = (half_t)v;
        }
    }
}

// ---------------- output projection + bias + residual ----------------
__global__ __launch_bounds__(256)
void proj_kernel(const half_t* __restrict__ o16, const half_t* __restrict__ wp16,
                 const float* __restrict__ bp, const float* __restrict__ x,
                 float* __restrict__ out) {
    int Mb = blockIdx.y * 128, Nb = blockIdx.x * 64;
    int w  = threadIdx.x >> 5;
    int wm = w >> 1, wn = w & 1;
    int mBase = Mb + wm * 32, nBase = Nb + wn * 32;

    v8f acc[2][2] = {};
    for (int kb = 0; kb < C_; kb += 32) {
        v16h a0 = load_a_frag(o16 + (size_t)mBase * C_ + kb, C_);
        v16h a1 = load_a_frag(o16 + (size_t)(mBase + 16) * C_ + kb, C_);
        v16h b0 = load_b_frag(wp16 + (size_t)nBase * C_ + kb, C_);
        v16h b1 = load_b_frag(wp16 + (size_t)(nBase + 16) * C_ + kb, C_);
        acc[0][0] = wmma_f16(a0, b0, acc[0][0]);
        acc[0][1] = wmma_f16(a0, b1, acc[0][1]);
        acc[1][0] = wmma_f16(a1, b0, acc[1][0]);
        acc[1][1] = wmma_f16(a1, b1, acc[1][1]);
    }

    int lane = threadIdx.x & 31, lr = lane & 15, lh = lane >> 4;
#pragma unroll
    for (int i = 0; i < 2; ++i)
#pragma unroll
        for (int j = 0; j < 2; ++j) {
            int col = nBase + j * 16 + lr;       // output channel c
            float bc = bp[col];
#pragma unroll
            for (int r = 0; r < 8; ++r) {
                int row = mBase + i * 16 + r + lh * 8;   // token b*1024+s
                int b = row >> 10, s = row & 1023;
                size_t addr = (size_t)b * C_ * N_ + (size_t)col * N_ + s;
                out[addr] = acc[i][j][r] + bc + x[addr];
            }
        }
}

// ---------------- host launch ----------------

extern "C" void kernel_launch(void* const* d_in, const int* in_sizes, int n_in,
                              void* d_out, int out_size, void* d_ws, size_t ws_size,
                              hipStream_t stream) {
    const float* x        = (const float*)d_in[0];
    const float* gn_scale = (const float*)d_in[1];
    const float* gn_bias  = (const float*)d_in[2];
    const float* wq = (const float*)d_in[3];
    const float* bq = (const float*)d_in[4];
    const float* wk = (const float*)d_in[5];
    const float* bk = (const float*)d_in[6];
    const float* wv = (const float*)d_in[7];
    const float* bv = (const float*)d_in[8];
    const float* wp = (const float*)d_in[9];
    const float* bp = (const float*)d_in[10];

    constexpr size_t MBYTE = 1024ull * 1024ull;
    constexpr size_t TOK_BYTES = (size_t)B_ * N_ * C_ * sizeof(half_t);  // 8 MB

    char* ws = (char*)d_ws;
    half_t* xn16 = (half_t*)(ws + 0 * TOK_BYTES);
    half_t* q16  = (half_t*)(ws + 1 * TOK_BYTES);
    half_t* k16  = (half_t*)(ws + 2 * TOK_BYTES);
    half_t* vt16 = (half_t*)(ws + 3 * TOK_BYTES);
    half_t* o16  = (half_t*)(ws + 4 * TOK_BYTES);
    half_t* w16  = (half_t*)(ws + 5 * TOK_BYTES);          // 4 x 512KB f16 weights
    float*  stats = (float*)(ws + 5 * TOK_BYTES + 2 * MBYTE);

    const int WN = C_ * C_;  // 262144
    cast_w_kernel<<<(WN + 255) / 256, 256, 0, stream>>>(wq, w16 + 0 * (size_t)WN, WN);
    cast_w_kernel<<<(WN + 255) / 256, 256, 0, stream>>>(wk, w16 + 1 * (size_t)WN, WN);
    cast_w_kernel<<<(WN + 255) / 256, 256, 0, stream>>>(wv, w16 + 2 * (size_t)WN, WN);
    cast_w_kernel<<<(WN + 255) / 256, 256, 0, stream>>>(wp, w16 + 3 * (size_t)WN, WN);

    gn_stats_kernel<<<B_ * 32, 256, 0, stream>>>(x, stats);
    gn_apply_kernel<<<(B_ * N_) / 256, 256, 0, stream>>>(x, gn_scale, gn_bias, stats, xn16);

    dim3 gqkv(C_ / 64, (B_ * N_) / 128, 3);
    qkv_kernel<<<gqkv, 256, 0, stream>>>(xn16, w16, bq, bk, bv, q16, k16, vt16);

    dim3 gattn(N_ / 128, B_ * HEADS_);
    attn_kernel<<<gattn, 256, 0, stream>>>(q16, k16, vt16, o16);

    dim3 gproj(C_ / 64, (B_ * N_) / 128);
    proj_kernel<<<gproj, 256, 0, stream>>>(o16, w16 + 3 * (size_t)WN, bp, x, (float*)d_out);
}
